// KernelNNBoundary_57775900066582
// MI455X (gfx1250) — matrix operand order repaired and checked by reading, not compiled, not run on hardware
//
#include <hip/hip_runtime.h>

typedef __attribute__((ext_vector_type(16))) __bf16 v16bf;
typedef __attribute__((ext_vector_type(8)))  float  v8f;
typedef __attribute__((ext_vector_type(4)))  int    v4i;

// round-half-up bf16 (1 add + shift); plenty for this pipeline
static __device__ __forceinline__ unsigned short f2bf(float f) {
  return (unsigned short)((__float_as_uint(f) + 0x8000u) >> 16);
}
static __device__ __forceinline__ float bf2f(unsigned short h) {
  return __uint_as_float(((unsigned)h) << 16);
}
// pack two f32 -> two bf16 in one dword (low = a, high = b): 2 adds + 1 v_perm
static __device__ __forceinline__ unsigned pk2bf(float a, float b) {
  unsigned ua = __float_as_uint(a) + 0x8000u;
  unsigned ub = __float_as_uint(b) + 0x8000u;
  return __builtin_amdgcn_perm(ub, ua, 0x07060302u);  // {ub.b3,ub.b2,ua.b3,ua.b2}
}

union ABf { unsigned short us[16]; v16bf v; };

#if __has_builtin(__builtin_amdgcn_global_load_async_to_lds_b128) && \
    __has_builtin(__builtin_amdgcn_s_wait_asynccnt)
#define USE_ASYNC_LDS 1
typedef __attribute__((address_space(1))) v4i* gv4p;   // global int32x4*
typedef __attribute__((address_space(3))) v4i* lv4p;   // LDS int32x4*
#else
#define USE_ASYNC_LDS 0
#endif

#define MSG_CHUNK 4

// ---------------------------------------------------------------- zero
__global__ __launch_bounds__(256) void zero_kernel(unsigned* p, long n) {
  long i = (long)blockIdx.x * blockDim.x + threadIdx.x;
  long st = (long)gridDim.x * blockDim.x;
  for (; i < n; i += st) p[i] = 0u;
}

// ------------------------------------------------- prep: swizzle fragments
// A-frag K map (16-bit A 16x32):  K = (lane>>4)*16 + i
// B-frag K map (16-bit B 32x16):  K = (lane>>4)*16 + i ; col = lane&15
// fragA/fragB hold w3^T tiles as WMMA *A* operands:
//   A[mrow,K] = w3[K, ct*16+mrow]  (mrow = gemmcol offset)
__global__ __launch_bounds__(256) void prep_kernel(
    const float* __restrict__ w3a, const float* __restrict__ w3b,
    const float* __restrict__ r1, const float* __restrict__ r2,
    const float* __restrict__ bi1, const float* __restrict__ bi2,
    unsigned short* __restrict__ fragA, unsigned short* __restrict__ fragB,
    unsigned short* __restrict__ rootfrag, float* __restrict__ biassum) {
  const int total = 32768 + 32768 + 1024 + 32;
  for (int idx = blockIdx.x * blockDim.x + threadIdx.x; idx < total;
       idx += gridDim.x * blockDim.x) {
    if (idx < 65536) {
      int f = idx & 32767;
      const float* w3 = (idx < 32768) ? w3a : w3b;
      unsigned short* dp = (idx < 32768) ? fragA : fragB;
      int ct = f >> 9, r = (f >> 4) & 31, i = f & 15;
      int K = ((r >> 4) << 4) + i;   // A-operand K map
      int mrow = r & 15;
      dp[f] = f2bf(w3[K * 1024 + ct * 16 + mrow]);
    } else if (idx < 66560) {
      int f = idx - 65536;                // rootsum = root1+root2 (B operand map)
      int ct = f >> 9, r = (f >> 4) & 31, i = f & 15;
      int K = ((r >> 4) << 4) + i;
      int col = ct * 16 + (r & 15);
      rootfrag[f] = f2bf(r1[K * 32 + col] + r2[K * 32 + col]);
    } else {
      int j = idx - 66560;
      biassum[j] = bi1[j] + bi2[j];
    }
  }
}

// ---------------------------------------------------------------- fc1
__global__ __launch_bounds__(256) void fc1_kernel(
    const float* __restrict__ x, const float* __restrict__ w,
    const float* __restrict__ b, float* __restrict__ h,
    unsigned short* __restrict__ hb, int N) {
  int i = blockIdx.x * 256 + threadIdx.x;
  if (i >= N * 32) return;
  int n = i >> 5, j = i & 31;
  float v = fmaf(x[n], w[j], b[j]);
  h[i] = v;
  hb[i] = f2bf(v);
}

// ------------------------------------------- small edge MLP: 6 -> 8 -> 32 (bf16 out)
__global__ __launch_bounds__(256) void edge_mlp_kernel(
    const float* __restrict__ ea, const float* __restrict__ w1,
    const float* __restrict__ b1, const float* __restrict__ w2,
    const float* __restrict__ b2, unsigned short* __restrict__ h2, int nE) {
  int e = blockIdx.x * 8 + (threadIdx.x >> 5);
  if (e >= nE) return;
  int lane = threadIdx.x & 31;
  const float* a = ea + (size_t)e * 6;
  float mid[8];
#pragma unroll
  for (int k = 0; k < 8; ++k) {
    float s = b1[k];
#pragma unroll
    for (int j = 0; j < 6; ++j) s = fmaf(a[j], w1[j * 8 + k], s);
    mid[k] = s > 0.f ? s : 0.f;
  }
  float o = b2[lane];
#pragma unroll
  for (int k = 0; k < 8; ++k) o = fmaf(mid[k], w2[k * 32 + lane], o);
  o = o > 0.f ? o : 0.f;
  h2[(size_t)e * 32 + lane] = f2bf(o);
}

// ------------- big GEMM: W_e = h2[e,:] x w3  ->  bf16 [nE,1024], row-major
// A = w3^T tile (reloaded per ct, 64KB L0/L2-hot), B = edge tile (registers,
// loop-invariant). D = [gemmcol x edge]: lane owns 8 consecutive outputs of one
// edge -> pack to 4 dwords, single b128 store per WMMA result.
__global__ __launch_bounds__(256) void edge_w_wmma(
    const unsigned short* __restrict__ h2, const unsigned short* __restrict__ afrag,
    const float* __restrict__ b3, unsigned short* __restrict__ Wout, int nE) {
  int nT = (nE + 31) >> 5;
  int tile = blockIdx.x * 8 + (threadIdx.x >> 5);
  if (tile >= nT) return;               // wave-uniform: EXEC stays all-ones
  bool full = ((int)(blockIdx.x + 1) * 256 <= nE);   // scalar (blockIdx-only)
  int lane = threadIdx.x & 31;
  int n = lane & 15;                    // local edge (D column)
  int khalf = lane >> 4;                // K half for B operand
  int base = tile << 5;
  int e0 = base + n;       if (e0 > nE - 1) e0 = nE - 1;
  int e1 = base + 16 + n;  if (e1 > nE - 1) e1 = nE - 1;
  // B fragment = 16 contiguous bf16 of the h2 row (K = khalf*16 + i)
  v16bf B0 = *(const v16bf*)(h2 + (size_t)e0 * 32 + khalf * 16);
  v16bf B1 = *(const v16bf*)(h2 + (size_t)e1 * 32 + khalf * 16);
  int rbase = (lane < 16) ? 0 : 8;      // D row half = gemmcol offset
  unsigned short* w0 = Wout + (size_t)(base + n) * 1024 + rbase;
  unsigned short* w1 = Wout + (size_t)(base + 16 + n) * 1024 + rbase;
  for (int ct = 0; ct < 64; ++ct) {
    v16bf av = *(const v16bf*)(afrag + ct * 512 + lane * 16);
    float4 blo = *(const float4*)(b3 + ct * 16 + rbase);
    float4 bhi = *(const float4*)(b3 + ct * 16 + rbase + 4);
    v8f c0 = {blo.x, blo.y, blo.z, blo.w, bhi.x, bhi.y, bhi.z, bhi.w};
    v8f c1 = c0;
    c0 = __builtin_amdgcn_wmma_f32_16x16x32_bf16(false, av, false, B0,
                                                 (short)0, c0, false, false);
    c1 = __builtin_amdgcn_wmma_f32_16x16x32_bf16(false, av, false, B1,
                                                 (short)0, c1, false, false);
    uint4 q0 = make_uint4(pk2bf(c0[0], c0[1]), pk2bf(c0[2], c0[3]),
                          pk2bf(c0[4], c0[5]), pk2bf(c0[6], c0[7]));
    uint4 q1 = make_uint4(pk2bf(c1[0], c1[1]), pk2bf(c1[2], c1[3]),
                          pk2bf(c1[4], c1[5]), pk2bf(c1[6], c1[7]));
    if (full) {
      *(uint4*)(w0 + ct * 16) = q0;
      *(uint4*)(w1 + ct * 16) = q1;
    } else {
      if (base + n < nE)      *(uint4*)(w0 + ct * 16) = q0;
      if (base + 16 + n < nE) *(uint4*)(w1 + ct * 16) = q1;
    }
  }
}

// ---------------------------------------------------------------- degree counts
__global__ __launch_bounds__(256) void count_kernel(const int* __restrict__ dst,
                                                    int* __restrict__ cnt, int nE) {
  int e = blockIdx.x * 256 + threadIdx.x;
  if (e < nE) atomicAdd(cnt + dst[e], 1);
}
__global__ __launch_bounds__(256) void inv_kernel(const int* __restrict__ cnt,
                                                  float* __restrict__ inv, int N) {
  int n = blockIdx.x * 256 + threadIdx.x;
  if (n < N) inv[n] = 1.0f / fmaxf((float)cnt[n], 1.0f);
}

// ------------------- per-edge matvec msg = h[src]^T * W_e  + scatter-add (f32)
// Each wave owns MSG_CHUNK edges; the 2KB per-edge matrix is double-buffered
// through LDS with async loads (ASYNCcnt); LDS indexed directly -> ds_load.
__global__ __launch_bounds__(256) void msg_kernel(
    const unsigned short* __restrict__ hb, const unsigned short* __restrict__ W,
    const int* __restrict__ src, const int* __restrict__ dst,
    float* __restrict__ agg, int nE) {
  int wv = threadIdx.x >> 5, lane = threadIdx.x & 31;
  int half = lane >> 4, mm = lane & 15;
  int e0 = (blockIdx.x * 8 + wv) * MSG_CHUNK;
  if (e0 >= nE) return;                 // wave-uniform
  int eend = e0 + MSG_CHUNK; if (eend > nE) eend = nE;
#if USE_ASYNC_LDS
  __shared__ unsigned lbuf32[8][2][512];   // 32KB: 2KB x 2 buffers per wave
  {
    const unsigned short* g = W + (size_t)e0 * 1024 + lane * 8;
    lv4p l0 = (lv4p)&lbuf32[wv][0][lane * 4];
    __builtin_amdgcn_global_load_async_to_lds_b128((gv4p)g, l0, 0, 0);
    __builtin_amdgcn_global_load_async_to_lds_b128((gv4p)g, l0, 512, 0);
    __builtin_amdgcn_global_load_async_to_lds_b128((gv4p)g, l0, 1024, 0);
    __builtin_amdgcn_global_load_async_to_lds_b128((gv4p)g, l0, 1536, 0);
  }
  for (int e = e0; e < eend; ++e) {
    int cur = (e - e0) & 1;
    if (e + 1 < eend) {                 // prefetch next edge matrix into alt buffer
      const unsigned short* g = W + (size_t)(e + 1) * 1024 + lane * 8;
      lv4p ln = (lv4p)&lbuf32[wv][cur ^ 1][lane * 4];
      __builtin_amdgcn_global_load_async_to_lds_b128((gv4p)g, ln, 0, 0);
      __builtin_amdgcn_global_load_async_to_lds_b128((gv4p)g, ln, 512, 0);
      __builtin_amdgcn_global_load_async_to_lds_b128((gv4p)g, ln, 1024, 0);
      __builtin_amdgcn_global_load_async_to_lds_b128((gv4p)g, ln, 1536, 0);
      __builtin_amdgcn_s_wait_asynccnt(4);   // in-order: current buffer complete
    } else {
      __builtin_amdgcn_s_wait_asynccnt(0);
    }
    int s = src[e], d = dst[e];
    float hval = bf2f(hb[(size_t)s * 32 + lane]);
    float a0 = 0.f, a1 = 0.f;
#pragma unroll
    for (int t = 0; t < 16; ++t) {
      int row = 2 * t + half;
      unsigned wvv = lbuf32[wv][cur][row * 16 + mm];   // ds_load, bank-conflict-free
      float hv = __shfl(hval, row, 32);
      a0 = fmaf(hv, __uint_as_float(wvv << 16), a0);
      a1 = fmaf(hv, __uint_as_float(wvv & 0xffff0000u), a1);
    }
    a0 += __shfl_xor(a0, 16, 32);
    a1 += __shfl_xor(a1, 16, 32);
    atomicAdd(agg + (size_t)d * 32 + (2 * mm + half), half ? a1 : a0);
  }
#else
  for (int e = e0; e < eend; ++e) {
    int s = src[e], d = dst[e];
    float hval = bf2f(hb[(size_t)s * 32 + lane]);
    const unsigned short* Wrow = W + (size_t)e * 1024;
    if (e + 1 < nE)
      __builtin_prefetch(W + (size_t)(e + 1) * 1024 + lane * 32, 0, 0);
    float a0 = 0.f, a1 = 0.f;
#pragma unroll
    for (int t = 0; t < 16; ++t) {
      int row = 2 * t + half;
      unsigned wvv = *(const unsigned*)(Wrow + row * 32 + mm * 2);
      float hv = __shfl(hval, row, 32);
      a0 = fmaf(hv, __uint_as_float(wvv << 16), a0);
      a1 = fmaf(hv, __uint_as_float(wvv & 0xffff0000u), a1);
    }
    a0 += __shfl_xor(a0, 16, 32);
    a1 += __shfl_xor(a1, 16, 32);
    atomicAdd(agg + (size_t)d * 32 + (2 * mm + half), half ? a1 : a0);
  }
#endif
}

// ------------ node update: relu(agg1/c1 + agg2/c2 + h@rootsum + biassum); re-zero agg
__global__ __launch_bounds__(256) void node_update_kernel(
    unsigned short* __restrict__ hb, const unsigned short* __restrict__ rootfrag,
    const float* __restrict__ biassum, float* __restrict__ agg1,
    float* __restrict__ agg2, const float* __restrict__ inv1,
    const float* __restrict__ inv2, float* __restrict__ h, int N) {
  int nT = (N + 15) >> 4;
  int tile = blockIdx.x * 8 + (threadIdx.x >> 5);
  if (tile >= nT) return;               // wave-uniform
  bool full = ((int)(blockIdx.x + 1) * 128 <= N);    // scalar (blockIdx-only)
  int lane = threadIdx.x & 31;
  int m = lane & 15, kb = (lane >> 4) << 3;
  int base = tile << 4;
  int arow = base + m; if (arow > N - 1) arow = N - 1;
  ABf A;
  const unsigned short* hrow = hb + (size_t)arow * 32;
#pragma unroll
  for (int i = 0; i < 16; ++i) A.us[i] = hrow[kb * 2 + i];
#pragma unroll
  for (int ct = 0; ct < 2; ++ct) {
    v16bf bv = *(const v16bf*)(rootfrag + ct * 512 + lane * 16);
    int col = ct * 16 + (lane & 15);
    float bias = biassum[col];
    v8f c = {bias, bias, bias, bias, bias, bias, bias, bias};
    c = __builtin_amdgcn_wmma_f32_16x16x32_bf16(false, A.v, false, bv,
                                                (short)0, c, false, false);
    int rbase = (lane < 16 ? 0 : 8);
#pragma unroll
    for (int r = 0; r < 8; ++r) {
      int row = base + rbase + r;
      if (full || row < N) {
        size_t o = (size_t)row * 32 + col;
        float v = c[r] + agg1[o] * inv1[row] + agg2[o] * inv2[row];
        v = v > 0.f ? v : 0.f;
        h[o] = v;
        hb[o] = f2bf(v);
        agg1[o] = 0.f;                  // ready for next depth iteration
        agg2[o] = 0.f;
      }
    }
  }
}

// ---------------------------------------------------------------- fc2: [N,32]->[N,1]
__global__ __launch_bounds__(256) void fc2_kernel(
    const float* __restrict__ h, const float* __restrict__ w,
    const float* __restrict__ b, float* __restrict__ out, int N) {
  int n = blockIdx.x * 8 + (threadIdx.x >> 5);
  if (n >= N) return;
  int lane = threadIdx.x & 31;
  float s = h[(size_t)n * 32 + lane] * w[lane];
#pragma unroll
  for (int dlt = 16; dlt > 0; dlt >>= 1) s += __shfl_xor(s, dlt, 32);
  if (lane == 0) out[n] = s + b[0];
}

extern "C" void kernel_launch(void* const* d_in, const int* in_sizes, int n_in,
                              void* d_out, int out_size, void* d_ws, size_t ws_size,
                              hipStream_t stream) {
  const float* x    = (const float*)d_in[0];
  const int*   ei   = (const int*)d_in[1];
  const float* ea   = (const float*)d_in[2];
  const int*   eib  = (const int*)d_in[3];
  const float* eab  = (const float*)d_in[4];
  const float* fc1w = (const float*)d_in[5];
  const float* fc1b = (const float*)d_in[6];
  const float* k1w1 = (const float*)d_in[7];
  const float* k1b1 = (const float*)d_in[8];
  const float* k1w2 = (const float*)d_in[9];
  const float* k1b2 = (const float*)d_in[10];
  const float* k1w3 = (const float*)d_in[11];
  const float* k1b3 = (const float*)d_in[12];
  const float* k2w1 = (const float*)d_in[13];
  const float* k2b1 = (const float*)d_in[14];
  const float* k2w2 = (const float*)d_in[15];
  const float* k2b2 = (const float*)d_in[16];
  const float* k2w3 = (const float*)d_in[17];
  const float* k2b3 = (const float*)d_in[18];
  const float* root1 = (const float*)d_in[19];
  const float* bias1 = (const float*)d_in[20];
  const float* root2 = (const float*)d_in[21];
  const float* bias2 = (const float*)d_in[22];
  const float* fc2w  = (const float*)d_in[23];
  const float* fc2b  = (const float*)d_in[24];
  (void)n_in; (void)out_size; (void)ws_size;

  const int N  = in_sizes[0];
  const int E  = in_sizes[1] / 2;
  const int EB = in_sizes[3] / 2;

  char* ws = (char*)d_ws;
  size_t off = 0;
  auto take = [&](size_t bytes) -> char* {
    char* p = ws + off;
    off = (off + bytes + 255) & ~(size_t)255;
    return p;
  };
  unsigned short* W1 = (unsigned short*)take((size_t)E * 1024 * 2);   // bf16 edge mats
  unsigned short* W2 = (unsigned short*)take((size_t)EB * 1024 * 2);
  unsigned short* h2 = (unsigned short*)take((size_t)(E + EB) * 32 * 2);
  float*          h  = (float*)take((size_t)N * 32 * 4);
  unsigned short* hb = (unsigned short*)take((size_t)N * 32 * 2);
  char* zbase = ws + off;                                             // zeroed region
  float* agg1 = (float*)take((size_t)N * 32 * 4);
  float* agg2 = (float*)take((size_t)N * 32 * 4);
  int*   cnt1 = (int*)take((size_t)N * 4);
  int*   cnt2 = (int*)take((size_t)N * 4);
  size_t zbytes = (size_t)((ws + off) - zbase);
  float* inv1 = (float*)take((size_t)N * 4);
  float* inv2 = (float*)take((size_t)N * 4);
  unsigned short* fragA    = (unsigned short*)take(32768 * 2);
  unsigned short* fragB    = (unsigned short*)take(32768 * 2);
  unsigned short* rootfrag = (unsigned short*)take(1024 * 2);
  float*          biassum  = (float*)take(32 * 4);

  zero_kernel<<<1024, 256, 0, stream>>>((unsigned*)zbase, (long)(zbytes / 4));
  prep_kernel<<<64, 256, 0, stream>>>(k1w3, k2w3, root1, root2, bias1, bias2,
                                      fragA, fragB, rootfrag, biassum);
  fc1_kernel<<<(N * 32 + 255) / 256, 256, 0, stream>>>(x, fc1w, fc1b, h, hb, N);
  edge_mlp_kernel<<<(E + 7) / 8, 256, 0, stream>>>(ea, k1w1, k1b1, k1w2, k1b2, h2, E);
  edge_mlp_kernel<<<(EB + 7) / 8, 256, 0, stream>>>(eab, k2w1, k2b1, k2w2, k2b2,
                                                    h2 + (size_t)E * 32, EB);
  int tE = (E + 31) / 32, tB = (EB + 31) / 32;
  edge_w_wmma<<<(tE + 7) / 8, 256, 0, stream>>>(h2, fragA, k1b3, W1, E);
  edge_w_wmma<<<(tB + 7) / 8, 256, 0, stream>>>(h2 + (size_t)E * 32, fragB, k2b3, W2, EB);
  count_kernel<<<(E + 255) / 256, 256, 0, stream>>>(ei + E, cnt1, E);
  count_kernel<<<(EB + 255) / 256, 256, 0, stream>>>(eib + EB, cnt2, EB);
  inv_kernel<<<(N + 255) / 256, 256, 0, stream>>>(cnt1, inv1, N);
  inv_kernel<<<(N + 255) / 256, 256, 0, stream>>>(cnt2, inv2, N);
  int nT = (N + 15) / 16;
  int mE = (E + 8 * MSG_CHUNK - 1) / (8 * MSG_CHUNK);
  int mB = (EB + 8 * MSG_CHUNK - 1) / (8 * MSG_CHUNK);
  for (int d = 0; d < 4; ++d) {
    msg_kernel<<<mE, 256, 0, stream>>>(hb, W1, ei, ei + E, agg1, E);
    msg_kernel<<<mB, 256, 0, stream>>>(hb, W2, eib, eib + EB, agg2, EB);
    node_update_kernel<<<(nT + 7) / 8, 256, 0, stream>>>(
        hb, rootfrag, biassum, agg1, agg2, inv1, inv2, h, N);
  }
  fc2_kernel<<<(N + 7) / 8, 256, 0, stream>>>(h, fc2w, fc2b, (float*)d_out, N);
}
